// PNA_22351009809233
// MI455X (gfx1250) — matrix operand order repaired and checked by reading, not compiled, not run on hardware
//
#include <hip/hip_runtime.h>
#include <hip/hip_bf16.h>

// ---------------------------------------------------------------------------
// PNA layer for MI455X (gfx1250), wave32 + v_wmma_f32_16x16x32_bf16.
//   K0: pack W_pre/W_post1/W_post2 (f32) into bf16 WMMA B-fragment layout.
//   K1: pretrans GEMM [16 nbrs x 144] @ [144 x 64] (2 nodes/wave, W_pre
//       fragments staged in LDS) + relu + mean/max/min/std aggregation.
//   K2: posttrans [32 nodes x 832] @ [832 x 64] + relu, then @ [64 x 64],
//       + bias + residual -> out (f32).  2 M-tiles/wave to amortize B loads.
// ---------------------------------------------------------------------------

typedef __bf16  v16bf __attribute__((ext_vector_type(16)));
typedef float   v8f   __attribute__((ext_vector_type(8)));

union Frag {
    v16bf    v;
    unsigned u[8];
    uint4    q[2];
};

#define LOG17  2.8332133440562162
#define AVGDL  2.302585092994046
#define S_AMP  ((float)(LOG17 / AVGDL))
#define S_ATT  ((float)(AVGDL / LOG17))

// A-matrix 16-bit K index for (vgpr v, lane-group g): interleaved halves.
__device__ __forceinline__ int kposA(int v, int g) {
    return (v < 4) ? (g * 8 + v * 2) : (16 + g * 8 + (v - 4) * 2);
}
// B-matrix 16-bit K index: interleaved halves, same order as kposA.
__device__ __forceinline__ int kposB(int v, int g) {
    return (v < 4) ? (g * 8 + v * 2) : (16 + g * 8 + (v - 4) * 2);
}

__device__ __forceinline__ unsigned bf16bits(float f) {
    __bf16 b = (__bf16)f;
    unsigned short u;
    __builtin_memcpy(&u, &b, 2);
    return (unsigned)u;
}

__device__ __forceinline__ float bfbits2f(unsigned us) {
    unsigned x = us << 16;
    float f;
    __builtin_memcpy(&f, &x, 4);
    return f;
}

// Convert float4 -> 4 x bf16, store as one 8-byte LDS write.
__device__ __forceinline__ void st_pk4(__bf16* dst, float4 f) {
    uint2 p;
    p.x = bf16bits(f.x) | (bf16bits(f.y) << 16);
    p.y = bf16bits(f.z) | (bf16bits(f.w) << 16);
    *(uint2*)dst = p;
}

// ---------------------------------------------------------------------------
// Kernel 0: pack weights into per-lane B-fragment dwords (bf16 pairs).
//   [0,5120)      W_pre   (K=144 -> 5 chunks, zero-padded to 160)
//   [5120,31744)  W_post1 (K=832 -> 26 chunks)
//   [31744,33792) W_post2 (K=64  -> 2 chunks)
// ---------------------------------------------------------------------------
__global__ void pna_prep(const float* __restrict__ Wpre,
                         const float* __restrict__ W1,
                         const float* __restrict__ W2,
                         unsigned*    __restrict__ frag)
{
    int t = blockIdx.x * 256 + threadIdx.x;
    if (t >= 33792) return;
    const float* W;
    int Kreal, local;
    if (t < 5120)       { W = Wpre; Kreal = 144; local = t; }
    else if (t < 31744) { W = W1;   Kreal = 832; local = t - 5120; }
    else                { W = W2;   Kreal = 64;  local = t - 31744; }

    int v    = local & 7;
    int lane = (local >> 3) & 31;
    int ct   = local >> 8;
    int kc   = ct >> 2;
    int nt   = ct & 3;
    int g    = lane >> 4;
    int n    = nt * 16 + (lane & 15);
    int k    = kc * 32 + kposB(v, g);

    float f0 = (k     < Kreal) ? W[k * 64 + n]       : 0.f;
    float f1 = (k + 1 < Kreal) ? W[(k + 1) * 64 + n] : 0.f;
    frag[t] = bf16bits(f0) | (bf16bits(f1) << 16);
}

// ---------------------------------------------------------------------------
// Kernel 1: pretrans + aggregation.  128 thr = 4 waves, 2 nodes/wave.
// ---------------------------------------------------------------------------
__global__ void __launch_bounds__(128)
pna_pretrans(const float* __restrict__ h,
             const float* __restrict__ edge,
             const int*   __restrict__ nbr,
             const float* __restrict__ b_pre,
             const unsigned* __restrict__ wpre_frag,
             __bf16*      __restrict__ aggbuf)
{
    __shared__ __bf16  zbuf[4][2][16][168];   // 43 KB: 2 nodes x 16 rows x 160K
    __shared__ unsigned wlds[5120];           // 20 KB: W_pre fragments

    const int wave = threadIdx.x >> 5;
    const int lane = threadIdx.x & 31;
    const int g    = lane >> 4;
    const int m    = lane & 15;
    const int nb   = blockIdx.x * 8 + wave * 2;   // first node of this wave

    // ---- stage W_pre fragments into LDS (coalesced uint4 copies) ----
    {
        uint4*       dst = (uint4*)wlds;
        const uint4* src = (const uint4*)wpre_frag;
        for (int i = 0; i < 10; ++i)
            dst[threadIdx.x + i * 128] = src[threadIdx.x + i * 128];
    }

    // ---- stage z = [h_src | h_dst | edge_attr | 0] as bf16 (vectorized) ----
    for (int j = 0; j < 2; ++j) {
        const int     n    = nb + j;
        const int     src  = nbr[n * 16 + m];
        const float4* s4   = (const float4*)(h + (size_t)src * 64);
        const float4* d4   = (const float4*)(h + (size_t)n * 64);
        const float4* e4   = (const float4*)(edge + ((size_t)n * 16 + m) * 16);
        __bf16*       row  = &zbuf[wave][j][m][0];
        if (g == 0) {                       // cols 0..79
            for (int q = 0; q < 16; ++q) st_pk4(row + q * 4, s4[q]);
            for (int q = 0; q < 4;  ++q) st_pk4(row + 64 + q * 4, d4[q]);
        } else {                            // cols 80..159
            for (int q = 4; q < 16; ++q) st_pk4(row + 64 + q * 4, d4[q]);
            for (int q = 0; q < 4;  ++q) st_pk4(row + 128 + q * 4, e4[q]);
            float4 z0 = {0.f, 0.f, 0.f, 0.f};
            for (int q = 0; q < 4;  ++q) st_pk4(row + 144 + q * 4, z0);
        }
    }
    __syncthreads();

    // ---- e = relu(z @ W_pre + b): 5 K-chunks x 4 N-tiles x 2 nodes ----
    v8f acc[2][4] = {};
    for (int kc = 0; kc < 5; ++kc) {
        Frag a[2];
#pragma unroll
        for (int j = 0; j < 2; ++j)
#pragma unroll
            for (int v = 0; v < 8; ++v) {
                int k = kc * 32 + kposA(v, g);
                a[j].u[v] = *(const unsigned*)&zbuf[wave][j][m][k];
            }
#pragma unroll
        for (int nt = 0; nt < 4; ++nt) {
            Frag b;
            const uint4* bp = (const uint4*)&wlds[((kc * 4 + nt) * 32 + lane) * 8];
            b.q[0] = bp[0];
            b.q[1] = bp[1];
#pragma unroll
            for (int j = 0; j < 2; ++j)
                acc[j][nt] = __builtin_amdgcn_wmma_f32_16x16x32_bf16(
                    false, a[j].v, false, b.v, (short)0, acc[j][nt], false, false);
        }
    }

    // ---- bias + relu + reduce over neighbors (8 VGPRs + lane^16) ----
#pragma unroll
    for (int j = 0; j < 2; ++j) {
        const int n = nb + j;
#pragma unroll
        for (int nt = 0; nt < 4; ++nt) {
            float bias = b_pre[nt * 16 + m];
            float s = 0.f, sq = 0.f, mx = -3.4e38f, mn = 3.4e38f;
#pragma unroll
            for (int v = 0; v < 8; ++v) {
                float e = acc[j][nt][v] + bias;
                e = e > 0.f ? e : 0.f;
                s  += e;
                sq += e * e;
                mx = fmaxf(mx, e);
                mn = fminf(mn, e);
            }
            s  += __shfl_xor(s, 16, 32);
            sq += __shfl_xor(sq, 16, 32);
            mx  = fmaxf(mx, __shfl_xor(mx, 16, 32));
            mn  = fminf(mn, __shfl_xor(mn, 16, 32));
            if (g == 0) {
                float mean = s * 0.0625f;
                float var  = sq * 0.0625f - mean * mean;
                var = var > 0.f ? var : 0.f;
                float sd = sqrtf(var + 1e-5f);
                __bf16* ab = aggbuf + (size_t)n * 256 + nt * 16 + m;
                ab[0]   = (__bf16)mean;
                ab[64]  = (__bf16)mx;
                ab[128] = (__bf16)mn;
                ab[192] = (__bf16)sd;
            }
        }
    }
}

// ---------------------------------------------------------------------------
// Kernel 2: posttrans.  32 thr = 1 wave, 2 M-tiles = 32 nodes per block.
// ---------------------------------------------------------------------------
__global__ void __launch_bounds__(32)
pna_posttrans(const float* __restrict__ h,
              const __bf16* __restrict__ aggbuf,
              const float* __restrict__ b1,
              const float* __restrict__ b2,
              const unsigned* __restrict__ w1_frag,
              const unsigned* __restrict__ w2_frag,
              float*       __restrict__ out)
{
    __shared__ __bf16 xs[32][840];         // 52.5 KB: 32 rows x 832 K (pad 840)
    __shared__ __bf16 rs[32][72];          // 4.5 KB: relu intermediate

    const int base = blockIdx.x * 32;
    const int lane = threadIdx.x & 31;
    const int g    = lane >> 4;
    const int m    = lane & 15;

    // ---- stage x = [h | agg | agg*S_AMP | agg*S_ATT] (vec4, coalesced) ----
    for (int i = 0; i < 208; ++i) {        // 32 rows x 208 vec4 per row
        int e  = i * 32 + lane;
        int r  = e / 208;
        int c  = (e % 208) * 4;
        float4 f;
        if (c < 64) {
            f = *(const float4*)(h + (size_t)(base + r) * 64 + c);
        } else {
            int a   = c - 64;
            int grp = a >> 8;
            float sc = (grp == 0) ? 1.f : ((grp == 1) ? S_AMP : S_ATT);
            uint2 p = *(const uint2*)(aggbuf + (size_t)(base + r) * 256 + (a & 255));
            f.x = bfbits2f(p.x & 0xffffu) * sc;
            f.y = bfbits2f(p.x >> 16)     * sc;
            f.z = bfbits2f(p.y & 0xffffu) * sc;
            f.w = bfbits2f(p.y >> 16)     * sc;
        }
        st_pk4(&xs[r][c], f);
    }
    __syncthreads();

    // ---- GEMM1: [32 x 832] @ [832 x 64], B shared across 2 M-tiles ----
    v8f acc[2][4] = {};
    for (int kc = 0; kc < 26; ++kc) {
        Frag a[2];
#pragma unroll
        for (int t = 0; t < 2; ++t)
#pragma unroll
            for (int v = 0; v < 8; ++v) {
                int k = kc * 32 + kposA(v, g);
                a[t].u[v] = *(const unsigned*)&xs[t * 16 + m][k];
            }
#pragma unroll
        for (int nt = 0; nt < 4; ++nt) {
            Frag b;
            const uint4* bp = (const uint4*)(w1_frag + ((kc * 4 + nt) * 32 + lane) * 8);
            b.q[0] = bp[0];
            b.q[1] = bp[1];
#pragma unroll
            for (int t = 0; t < 2; ++t)
                acc[t][nt] = __builtin_amdgcn_wmma_f32_16x16x32_bf16(
                    false, a[t].v, false, b.v, (short)0, acc[t][nt], false, false);
        }
    }

    // ---- bias + relu, re-layout C->A through LDS ----
#pragma unroll
    for (int t = 0; t < 2; ++t)
#pragma unroll
        for (int nt = 0; nt < 4; ++nt) {
            float bias = b1[nt * 16 + m];
#pragma unroll
            for (int v = 0; v < 8; ++v) {
                float e = acc[t][nt][v] + bias;
                e = e > 0.f ? e : 0.f;
                rs[t * 16 + v + 8 * g][nt * 16 + m] = (__bf16)e;
            }
        }
    __syncthreads();

    // ---- GEMM2: [32 x 64] @ [64 x 64] ----
    v8f acc2[2][4] = {};
    for (int kc = 0; kc < 2; ++kc) {
        Frag a[2];
#pragma unroll
        for (int t = 0; t < 2; ++t)
#pragma unroll
            for (int v = 0; v < 8; ++v) {
                int k = kc * 32 + kposA(v, g);
                a[t].u[v] = *(const unsigned*)&rs[t * 16 + m][k];
            }
#pragma unroll
        for (int nt = 0; nt < 4; ++nt) {
            Frag b;
            const uint4* bp = (const uint4*)(w2_frag + ((kc * 4 + nt) * 32 + lane) * 8);
            b.q[0] = bp[0];
            b.q[1] = bp[1];
#pragma unroll
            for (int t = 0; t < 2; ++t)
                acc2[t][nt] = __builtin_amdgcn_wmma_f32_16x16x32_bf16(
                    false, a[t].v, false, b.v, (short)0, acc2[t][nt], false, false);
        }
    }

    // ---- bias + residual, store f32 ----
#pragma unroll
    for (int t = 0; t < 2; ++t)
#pragma unroll
        for (int nt = 0; nt < 4; ++nt) {
            int   N    = nt * 16 + m;
            float bias = b2[N];
#pragma unroll
            for (int v = 0; v < 8; ++v) {
                int node = base + t * 16 + v + 8 * g;
                out[(size_t)node * 64 + N] = acc2[t][nt][v] + bias + h[(size_t)node * 64 + N];
            }
        }
}

// ---------------------------------------------------------------------------
// Host launcher.  ws layout: [aggbuf bf16 N*256 = 32 MB][frag dwords 132 KB]
// ---------------------------------------------------------------------------
extern "C" void kernel_launch(void* const* d_in, const int* in_sizes, int n_in,
                              void* d_out, int out_size, void* d_ws, size_t ws_size,
                              hipStream_t stream)
{
    const float* h    = (const float*)d_in[0];
    const float* edge = (const float*)d_in[1];
    const int*   nbr  = (const int*)  d_in[2];
    const float* Wpre = (const float*)d_in[3];
    const float* bpre = (const float*)d_in[4];
    const float* W1   = (const float*)d_in[5];
    const float* b1   = (const float*)d_in[6];
    const float* W2   = (const float*)d_in[7];
    const float* b2   = (const float*)d_in[8];
    float*       out  = (float*)d_out;

    __bf16*   aggbuf = (__bf16*)d_ws;
    unsigned* frag   = (unsigned*)((char*)d_ws + (size_t)65536 * 256 * 2);

    pna_prep     <<<132,  256, 0, stream>>>(Wpre, W1, W2, frag);
    pna_pretrans <<<8192, 128, 0, stream>>>(h, edge, nbr, bpre, frag, aggbuf);
    pna_posttrans<<<2048,  32, 0, stream>>>(h, aggbuf, b1, b2,
                                            frag + 5120, frag + 31744, out);
}
